// DiffPoolStep_28312424415602
// MI455X (gfx1250) — matrix operand order repaired and checked by reading, not compiled, not run on hardware
//
#include <hip/hip_runtime.h>
#include <hip/hip_bf16.h>
#include <math.h>

// ---------------------------------------------------------------------------
// DiffPool step for MI455X (gfx1250).
// All large matmuls: v_wmma_f32_16x16x32_bf16 (bf16 operands - exact for the
// binary adjacency - with fp32 accumulation).
// GEMM kernels use an LDS double-buffered B-panel pipeline: all 8 waves of a
// block share one 128col x 32k panel. The next panel is fetched with the
// CDNA5 async copy path (global_load_async_to_lds_b128, ASYNCcnt) issued
// before the WMMA burst, published with s_wait_asynccnt + split barrier, so
// L2 latency overlaps matrix math with no register round-trip.
// ---------------------------------------------------------------------------

#define BB 8
#define NN 2048
#define DD 128
#define LDS_PITCH 40   // 32 k-elements + 8 pad (80B rows) to spread LDS banks
#define USE_ASYNC_LDS 1

typedef __bf16 bf16;
typedef __attribute__((ext_vector_type(16))) __bf16 bf16x16;
typedef __attribute__((ext_vector_type(8)))  __bf16 bf16x8;
typedef __attribute__((ext_vector_type(8)))  float  f32x8;

#define DEVINL __device__ __forceinline__

DEVINL f32x8 splat8(float v) {
  f32x8 r;
#pragma unroll
  for (int i = 0; i < 8; ++i) r[i] = v;
  return r;
}

DEVINL bf16x16 combine16(bf16x8 lo, bf16x8 hi) {
  bf16x16 r;
#pragma unroll
  for (int e = 0; e < 8; ++e) { r[e] = lo[e]; r[e + 8] = hi[e]; }
  return r;
}

DEVINL f32x8 wmma_bf16(bf16x16 a, bf16x16 b, f32x8 c) {
  return __builtin_amdgcn_wmma_f32_16x16x32_bf16(false, a, false, b,
                                                 (short)0, c, false, false);
}

DEVINL void wait_async0() {
#if __has_builtin(__builtin_amdgcn_s_wait_asynccnt)
  __builtin_amdgcn_s_wait_asynccnt(0);
#else
  asm volatile("s_wait_asynccnt 0x0" ::: "memory");
#endif
}

// B fragment from the LDS panel: col = t*16 + lr, k-halves per lane half.
// (B operand 32x16: lanes 0-15 carry K {0..7,16..23}, lanes 16-31 carry K {8..15,24..31}.)
DEVINL bf16x16 lds_fragB(const bf16* buf, int t, int lr, int half) {
  const bf16* p = buf + (size_t)(t * 16 + lr) * LDS_PITCH + half * 8;
  bf16x8 lo = *reinterpret_cast<const bf16x8*>(p);
  bf16x8 hi = *reinterpret_cast<const bf16x8*>(p + 16);
  return combine16(lo, hi);
}

// ---- panel staging: 256 threads x 32B = 128 cols x 32 ks (bf16) ----------
#if USE_ASYNC_LDS
// CDNA5 async copy: per-lane global -> LDS, tracked by ASYNCcnt. INST_OFFSET
// applies to both the global and LDS side, so one address pair + offset:16
// moves the full 32B chunk.
DEVINL void stage_async(const bf16* __restrict__ Bcol0, size_t ldb, int k0,
                        bf16* dstbuf, int tid) {
  const bf16* src = Bcol0 + (size_t)(tid >> 1) * ldb + k0 + (tid & 1) * 16;
  bf16* dst = dstbuf + (size_t)(tid >> 1) * LDS_PITCH + (tid & 1) * 16;
  unsigned lds_off = (unsigned)(size_t)dst;       // low 32b = LDS byte offset
  unsigned long long ga = (unsigned long long)(size_t)src;
  asm volatile("global_load_async_to_lds_b128 %0, %1, off\n\t"
               "global_load_async_to_lds_b128 %0, %1, off offset:16"
               :: "v"(lds_off), "v"(ga) : "memory");
}
#else
DEVINL void stage_fill(bf16x8& s0, bf16x8& s1, const bf16* __restrict__ Bcol0,
                       size_t ldb, int k0, int tid) {
  const bf16* p = Bcol0 + (size_t)(tid >> 1) * ldb + k0 + (tid & 1) * 16;
  s0 = *reinterpret_cast<const bf16x8*>(p);
  s1 = *reinterpret_cast<const bf16x8*>(p + 8);
}
DEVINL void stage_store(bf16* buf, bf16x8 s0, bf16x8 s1, int tid) {
  bf16* p = buf + (size_t)(tid >> 1) * LDS_PITCH + (tid & 1) * 16;
  *reinterpret_cast<bf16x8*>(p) = s0;
  *reinterpret_cast<bf16x8*>(p + 8) = s1;
}
#endif

// Pipelined 16x128 output strip GEMM: acc += A[16xK] * B[Kx128].
// Arow: this lane's A row (A stored [M,K] row-major; lane layout ISA 7.12.2:
// lanes 0-15 K {0..7,16..23}, lanes 16-31 K {8..15,24..31}).
// Bcol0: base of shared B^T panel rows ([col,K] row-major, ld = ldb).
// lds0/lds1: two 128*LDS_PITCH bf16 buffers. Whole block must call this.
DEVINL void gemm_panel(const bf16* __restrict__ Arow, const bf16* __restrict__ Bcol0,
                       size_t ldb, int K, bf16* lds0, bf16* lds1,
                       int tid, int lane, f32x8 acc[8]) {
  const int lr = lane & 15, half = lane >> 4;

#if USE_ASYNC_LDS
  // prologue: async-stage panel 0, prefetch A frag 0
  stage_async(Bcol0, ldb, 0, lds0, tid);
  bf16x8 alo = *reinterpret_cast<const bf16x8*>(Arow + half * 8);
  bf16x8 ahi = *reinterpret_cast<const bf16x8*>(Arow + 16 + half * 8);
  wait_async0();
  __syncthreads();

  for (int k0 = 0, it = 0; k0 < K; k0 += 32, ++it) {
    bf16* cur = (it & 1) ? lds1 : lds0;
    bf16* nxt = (it & 1) ? lds0 : lds1;
    const bool more = (k0 + 32) < K;

    // Issue next panel async copy + next A-frag load now; they complete
    // while the WMMA burst below runs.
    bf16x8 nlo, nhi;
    if (more) {
      stage_async(Bcol0, ldb, k0 + 32, nxt, tid);
      nlo = *reinterpret_cast<const bf16x8*>(Arow + k0 + 32 + half * 8);
      nhi = *reinterpret_cast<const bf16x8*>(Arow + k0 + 48 + half * 8);
    }
    __builtin_amdgcn_sched_barrier(0);  // keep issue above the WMMAs

    bf16x16 fa = combine16(alo, ahi);
#pragma unroll
    for (int t = 0; t < 8; ++t) {
      bf16x16 fb = lds_fragB(cur, t, lr, half);
      acc[t] = wmma_bf16(fa, fb, acc[t]);
    }

    if (more) { alo = nlo; ahi = nhi; }
    wait_async0();      // next panel fully in LDS before publishing
    __syncthreads();
  }
#else
  bf16x8 s0, s1;
  stage_fill(s0, s1, Bcol0, ldb, 0, tid);
  stage_store(lds0, s0, s1, tid);
  bf16x8 alo = *reinterpret_cast<const bf16x8*>(Arow + half * 8);
  bf16x8 ahi = *reinterpret_cast<const bf16x8*>(Arow + 16 + half * 8);
  __syncthreads();

  for (int k0 = 0, it = 0; k0 < K; k0 += 32, ++it) {
    bf16* cur = (it & 1) ? lds1 : lds0;
    bf16* nxt = (it & 1) ? lds0 : lds1;
    const bool more = (k0 + 32) < K;
    bf16x8 nlo, nhi;
    if (more) {
      stage_fill(s0, s1, Bcol0, ldb, k0 + 32, tid);
      nlo = *reinterpret_cast<const bf16x8*>(Arow + k0 + 32 + half * 8);
      nhi = *reinterpret_cast<const bf16x8*>(Arow + k0 + 48 + half * 8);
    }
    __builtin_amdgcn_sched_barrier(0);
    bf16x16 fa = combine16(alo, ahi);
#pragma unroll
    for (int t = 0; t < 8; ++t) {
      bf16x16 fb = lds_fragB(cur, t, lr, half);
      acc[t] = wmma_bf16(fa, fb, acc[t]);
    }
    if (more) {
      stage_store(nxt, s0, s1, tid);
      alo = nlo;
      ahi = nhi;
    }
    __syncthreads();
  }
#endif
}

// C/D 16x16 fp32 layout: lane L holds col n = L&15, rows r + 8*(L>>4), r=0..7.

// ---------------------------------------------------------------------------
// Prep kernels
// ---------------------------------------------------------------------------

__global__ void k_zero2(float* p) {
  if (threadIdx.x < 2) p[threadIdx.x] = 0.f;
}

__global__ void k_cvt_bf16(const float* __restrict__ in, bf16* __restrict__ out, int n) {
  int i = blockIdx.x * 256 + threadIdx.x;
  if (i < n) out[i] = (bf16)in[i];
}

// in-degree of target j: deg[b,j] = sum_i adj[b,i,j]; store 1/max(deg,1)
__global__ void k_deg(const float* __restrict__ adj, float* __restrict__ dinv) {
  int b = blockIdx.y;
  int j = blockIdx.x * 256 + threadIdx.x;
  const float* p = adj + (size_t)b * NN * NN + j;
  float s = 0.f;
  for (int i = 0; i < NN; ++i) s += p[(size_t)i * NN];
  dinv[b * NN + j] = 1.f / fmaxf(s, 1.f);
}

// adjT[b,j,i] = (bf16) adj[b,i,j]
__global__ void k_adjT(const float* __restrict__ adj, bf16* __restrict__ adjT) {
  __shared__ float tile[32][33];
  int b = blockIdx.z;
  int i0 = blockIdx.y * 32, j0 = blockIdx.x * 32;
  const float* in = adj + (size_t)b * NN * NN;
  bf16* out = adjT + (size_t)b * NN * NN;
  int tx = threadIdx.x, ty = threadIdx.y;  // 32 x 8
#pragma unroll
  for (int k = 0; k < 4; ++k)
    tile[ty + k * 8][tx] = in[(size_t)(i0 + ty + k * 8) * NN + j0 + tx];
  __syncthreads();
#pragma unroll
  for (int k = 0; k < 4; ++k)
    out[(size_t)(j0 + ty + k * 8) * NN + i0 + tx] = (bf16)tile[tx][ty + k * 8];
}

// [R,C] bf16 -> [C,R] bf16, per-batch stride R*C (grid.z = batch)
__global__ void k_trans_bf(const bf16* __restrict__ in, bf16* __restrict__ out,
                           int R, int C) {
  __shared__ bf16 tile[32][33];
  int b = blockIdx.z;
  in += (size_t)b * R * C;
  out += (size_t)b * R * C;
  int r0 = blockIdx.y * 32, c0 = blockIdx.x * 32;
  int tx = threadIdx.x, ty = threadIdx.y;
#pragma unroll
  for (int k = 0; k < 4; ++k)
    tile[ty + k * 8][tx] = in[(size_t)(r0 + ty + k * 8) * C + c0 + tx];
  __syncthreads();
#pragma unroll
  for (int k = 0; k < 4; ++k)
    out[(size_t)(c0 + ty + k * 8) * R + r0 + tx] = tile[tx][ty + k * 8];
}

// ---------------------------------------------------------------------------
// SAGE aggregation: agg[b,j,d] = dinv[b,j] * sum_i adjT[b,j,i] * xT[b,d,i]
// Block = 8 waves = 8 row tiles sharing the xT panel; K = 2048.
// ---------------------------------------------------------------------------
__global__ void k_sage_agg(const bf16* __restrict__ adjT, const bf16* __restrict__ xT,
                           const float* __restrict__ dinv, bf16* __restrict__ agg) {
  __shared__ __align__(32) bf16 lds[2][128 * LDS_PITCH];
  const int b = blockIdx.y;
  const int tid = threadIdx.x;
  const int wave = tid >> 5, lane = tid & 31;
  const int lr = lane & 15, half = lane >> 4;
  const int n0 = (blockIdx.x * 8 + wave) * 16;  // 0..2032

  const bf16* Arow = adjT + (size_t)b * NN * NN + (size_t)(n0 + lr) * NN;
  const bf16* Bb = xT + (size_t)b * DD * NN;

  f32x8 acc[8];
#pragma unroll
  for (int t = 0; t < 8; ++t) acc[t] = splat8(0.f);

  gemm_panel(Arow, Bb, NN, NN, lds[0], lds[1], tid, lane, acc);

  const float* dv = dinv + (size_t)b * NN + n0 + half * 8;
  bf16* out = agg + (size_t)b * NN * DD;
#pragma unroll
  for (int r = 0; r < 8; ++r) {
    float s = dv[r];
    int row = n0 + half * 8 + r;
#pragma unroll
    for (int t = 0; t < 8; ++t)
      out[(size_t)row * DD + t * 16 + lr] = (bf16)(acc[t][r] * s);
  }
}

// ---------------------------------------------------------------------------
// SAGE linear + ReLU + row L2-norm:
//   h = agg @ Wl^T + bl + xin @ Wr^T ; h = relu(h); h /= (||h||_2 + 1e-9)
// Wl/Wr stored [o, d] row-major (K=d contiguous). K = 128 per GEMM.
// ---------------------------------------------------------------------------
__global__ void k_sage_lin(const bf16* __restrict__ aggb, const bf16* __restrict__ xinb,
                           const bf16* __restrict__ Wl, const float* __restrict__ bl,
                           const bf16* __restrict__ Wr,
                           bf16* __restrict__ outb, float* __restrict__ outf) {
  __shared__ __align__(32) bf16 lds[2][128 * LDS_PITCH];
  const int b = blockIdx.y;
  const int tid = threadIdx.x;
  const int wave = tid >> 5, lane = tid & 31;
  const int lr = lane & 15, half = lane >> 4;
  const int n0 = (blockIdx.x * 8 + wave) * 16;

  const bf16* Ar = aggb + (size_t)b * NN * DD + (size_t)(n0 + lr) * DD;
  const bf16* Xr = xinb + (size_t)b * NN * DD + (size_t)(n0 + lr) * DD;

  f32x8 acc[8];
#pragma unroll
  for (int t = 0; t < 8; ++t) acc[t] = splat8(bl[t * 16 + lr]);

  gemm_panel(Ar, Wl, DD, DD, lds[0], lds[1], tid, lane, acc);
  gemm_panel(Xr, Wr, DD, DD, lds[0], lds[1], tid, lane, acc);

  // ReLU then row L2 norm (row r+8*half lives in one 16-lane half)
#pragma unroll
  for (int t = 0; t < 8; ++t)
#pragma unroll
    for (int r = 0; r < 8; ++r) acc[t][r] = fmaxf(acc[t][r], 0.f);

  float inv[8];
#pragma unroll
  for (int r = 0; r < 8; ++r) {
    float s = 0.f;
#pragma unroll
    for (int t = 0; t < 8; ++t) s += acc[t][r] * acc[t][r];
    s += __shfl_xor(s, 1, 32);
    s += __shfl_xor(s, 2, 32);
    s += __shfl_xor(s, 4, 32);
    s += __shfl_xor(s, 8, 32);
    inv[r] = 1.f / (sqrtf(s) + 1e-9f);
  }

  bf16* ob = outb + (size_t)b * NN * DD;
  float* of = outf ? outf + (size_t)b * NN * DD : nullptr;
#pragma unroll
  for (int r = 0; r < 8; ++r) {
    int row = n0 + half * 8 + r;
#pragma unroll
    for (int t = 0; t < 8; ++t) {
      float v = acc[t][r] * inv[r];
      ob[(size_t)row * DD + t * 16 + lr] = (bf16)v;
      if (of) of[(size_t)row * DD + t * 16 + lr] = v;
    }
  }
}

// ---------------------------------------------------------------------------
// Softmax over last dim (128) + entropy accumulation. One wave per row.
// ---------------------------------------------------------------------------
__global__ void k_softmax(const float* __restrict__ a2, bf16* __restrict__ Sb,
                          float* __restrict__ ent_accum) {
  const int wave = threadIdx.x >> 5, lane = threadIdx.x & 31;
  const int row = blockIdx.x * 8 + wave;  // 0 .. B*N-1
  const float* p = a2 + (size_t)row * DD;

  float v[4];
#pragma unroll
  for (int j = 0; j < 4; ++j) v[j] = p[lane + 32 * j];
  float m = fmaxf(fmaxf(v[0], v[1]), fmaxf(v[2], v[3]));
#pragma unroll
  for (int d = 16; d >= 1; d >>= 1) m = fmaxf(m, __shfl_xor(m, d, 32));
  float e[4], s = 0.f;
#pragma unroll
  for (int j = 0; j < 4; ++j) { e[j] = expf(v[j] - m); s += e[j]; }
#pragma unroll
  for (int d = 16; d >= 1; d >>= 1) s += __shfl_xor(s, d, 32);
  float inv = 1.f / s;

  float ent = 0.f;
  bf16* o = Sb + (size_t)row * DD;
#pragma unroll
  for (int j = 0; j < 4; ++j) {
    float sv = e[j] * inv;
    o[lane + 32 * j] = (bf16)sv;
    ent -= sv * logf(sv + 1e-15f);
  }
#pragma unroll
  for (int d = 16; d >= 1; d >>= 1) ent += __shfl_xor(ent, d, 32);
  if (lane == 0) atomicAdd(ent_accum, ent);
}

// ---------------------------------------------------------------------------
// 128x128 output GEMM with K = 2048: C[r,c] = sum_k A[r,k] * Bt[c,k].
// Used for h_pooled (A=S^T, Bt=h_emb^T) and adj_pooled (A=P, Bt=S^T).
// ---------------------------------------------------------------------------
__global__ void k_gemm128(const bf16* __restrict__ Ab, const bf16* __restrict__ Bb,
                          float* __restrict__ out) {
  __shared__ __align__(32) bf16 lds[2][128 * LDS_PITCH];
  const int b = blockIdx.x;
  const int tid = threadIdx.x;
  const int wave = tid >> 5, lane = tid & 31;
  const int lr = lane & 15, half = lane >> 4;
  const int r0 = wave * 16;

  const bf16* Arow = Ab + (size_t)b * DD * NN + (size_t)(r0 + lr) * NN;
  const bf16* Bbat = Bb + (size_t)b * DD * NN;

  f32x8 acc[8];
#pragma unroll
  for (int t = 0; t < 8; ++t) acc[t] = splat8(0.f);

  gemm_panel(Arow, Bbat, NN, NN, lds[0], lds[1], tid, lane, acc);

  float* o = out + (size_t)b * DD * DD;
#pragma unroll
  for (int r = 0; r < 8; ++r) {
    int row = r0 + half * 8 + r;
#pragma unroll
    for (int t = 0; t < 8; ++t) o[(size_t)row * DD + t * 16 + lr] = acc[t][r];
  }
}

// P[b,k,m] = sum_n ST[b,k,n] * adjT[b,m,n]   (= S^T adj), bf16 out, ld = N
__global__ void k_pool_p(const bf16* __restrict__ ST, const bf16* __restrict__ adjT,
                         bf16* __restrict__ P) {
  __shared__ __align__(32) bf16 lds[2][128 * LDS_PITCH];
  const int b = blockIdx.y;
  const int tid = threadIdx.x;
  const int wave = tid >> 5, lane = tid & 31;
  const int lr = lane & 15, half = lane >> 4;
  const int r0 = wave * 16;            // k tile 0..7
  const int m0 = blockIdx.x * 128;     // column strip

  const bf16* Arow = ST + (size_t)b * DD * NN + (size_t)(r0 + lr) * NN;
  const bf16* Bcol = adjT + (size_t)b * NN * NN + (size_t)m0 * NN;

  f32x8 acc[8];
#pragma unroll
  for (int t = 0; t < 8; ++t) acc[t] = splat8(0.f);

  gemm_panel(Arow, Bcol, NN, NN, lds[0], lds[1], tid, lane, acc);

  bf16* o = P + (size_t)b * DD * NN;
#pragma unroll
  for (int r = 0; r < 8; ++r) {
    int row = r0 + half * 8 + r;
#pragma unroll
    for (int t = 0; t < 8; ++t)
      o[(size_t)row * NN + m0 + t * 16 + lr] = (bf16)acc[t][r];
  }
}

// link loss: sum over all (b,n,m) of (adj[b,n,m] - (S S^T)[b,n,m])^2
// Block = one 128-col strip (shared B panel), waves = 8 row tiles.
__global__ void k_link(const bf16* __restrict__ Sb, const float* __restrict__ adj,
                       float* __restrict__ accum) {
  __shared__ __align__(32) bf16 lds[2][128 * LDS_PITCH];
  const int b = blockIdx.z;
  const int tid = threadIdx.x;
  const int wave = tid >> 5, lane = tid & 31;
  const int lr = lane & 15, half = lane >> 4;
  const int n0 = (blockIdx.y * 8 + wave) * 16;
  const int m0 = blockIdx.x * 128;

  const bf16* Sbat = Sb + (size_t)b * NN * DD;
  const bf16* Arow = Sbat + (size_t)(n0 + lr) * DD;
  const bf16* Bcol = Sbat + (size_t)m0 * DD;

  f32x8 acc[8];
#pragma unroll
  for (int t = 0; t < 8; ++t) acc[t] = splat8(0.f);

  gemm_panel(Arow, Bcol, DD, DD, lds[0], lds[1], tid, lane, acc);

  const float* adjb = adj + (size_t)b * NN * NN;
  float local = 0.f;
#pragma unroll
  for (int r = 0; r < 8; ++r) {
    int row = n0 + half * 8 + r;
#pragma unroll
    for (int t = 0; t < 8; ++t) {
      float d = adjb[(size_t)row * NN + m0 + t * 16 + lr] - acc[t][r];
      local += d * d;
    }
  }
#pragma unroll
  for (int d = 16; d >= 1; d >>= 1) local += __shfl_xor(local, d, 32);
  if (lane == 0) atomicAdd(accum, local);
}

__global__ void k_final(const float* __restrict__ accum, float* __restrict__ out) {
  // link_loss = ||link||_F / adj.size ; ent_loss = ent_sum / (b*N)
  out[2 * BB * DD * DD + 0] = sqrtf(accum[0]) / (float)((size_t)BB * NN * NN);
  out[2 * BB * DD * DD + 1] = accum[1] / (float)(BB * NN);
}

// ---------------------------------------------------------------------------
// Host launcher
// ---------------------------------------------------------------------------
extern "C" void kernel_launch(void* const* d_in, const int* in_sizes, int n_in,
                              void* d_out, int out_size, void* d_ws, size_t ws_size,
                              hipStream_t stream) {
  (void)in_sizes; (void)n_in; (void)out_size; (void)ws_size;

  const float* x    = (const float*)d_in[0];
  const float* adj  = (const float*)d_in[1];
  const float* We1l = (const float*)d_in[2];
  const float* be1  = (const float*)d_in[3];
  const float* We1r = (const float*)d_in[4];
  const float* We2l = (const float*)d_in[5];
  const float* be2  = (const float*)d_in[6];
  const float* We2r = (const float*)d_in[7];
  const float* Wa1l = (const float*)d_in[8];
  const float* ba1  = (const float*)d_in[9];
  const float* Wa1r = (const float*)d_in[10];
  const float* Wa2l = (const float*)d_in[11];
  const float* ba2  = (const float*)d_in[12];
  const float* Wa2r = (const float*)d_in[13];
  float* out = (float*)d_out;

  char* w = (char*)d_ws;
  size_t off = 0;
  auto take = [&](size_t bytes) -> char* {
    char* p = w + off;
    off += (bytes + 255) & ~(size_t)255;
    return p;
  };

  const size_t FEAT = (size_t)BB * NN * DD;  // 2M elements
  bf16* adjT   = (bf16*)take((size_t)BB * NN * NN * 2);  // 64 MB
  bf16* xT     = (bf16*)take(FEAT * 2);
  bf16* xbf    = (bf16*)take(FEAT * 2);
  bf16* h1bf   = (bf16*)take(FEAT * 2);
  bf16* h1T    = (bf16*)take(FEAT * 2);
  bf16* hembbf = (bf16*)take(FEAT * 2);
  bf16* hembT  = (bf16*)take(FEAT * 2);
  bf16* a1bf   = (bf16*)take(FEAT * 2);
  bf16* a1T    = (bf16*)take(FEAT * 2);
  bf16* a2bf   = (bf16*)take(FEAT * 2);
  bf16* Sbf    = (bf16*)take(FEAT * 2);
  bf16* ST     = (bf16*)take(FEAT * 2);
  bf16* aggE1  = (bf16*)take(FEAT * 2);
  bf16* aggT   = (bf16*)take(FEAT * 2);
  bf16* Pbf    = (bf16*)take(FEAT * 2);
  float* a2f   = (float*)take(FEAT * 4);
  float* dinv  = (float*)take((size_t)BB * NN * 4);
  bf16* Wb[8];
  for (int i = 0; i < 8; ++i) Wb[i] = (bf16*)take((size_t)DD * DD * 2);
  float* accum = (float*)take(256);

  const dim3 blk256(256), blkT(32, 8);
  const dim3 gFeatT(DD / 32, NN / 32, BB);   // [N,D] -> [D,N] transpose
  const dim3 gStrip(16, BB);                 // 128 row tiles / 8 waves

  // --- prep ---
  k_zero2<<<1, 32, 0, stream>>>(accum);
  k_cvt_bf16<<<(int)((FEAT + 255) / 256), blk256, 0, stream>>>(x, xbf, (int)FEAT);
  const float* Wsrc[8] = {We1l, We1r, We2l, We2r, Wa1l, Wa1r, Wa2l, Wa2r};
  for (int i = 0; i < 8; ++i)
    k_cvt_bf16<<<64, blk256, 0, stream>>>(Wsrc[i], Wb[i], DD * DD);
  k_deg<<<dim3(NN / 256, BB), blk256, 0, stream>>>(adj, dinv);
  k_adjT<<<dim3(NN / 32, NN / 32, BB), blkT, 0, stream>>>(adj, adjT);
  k_trans_bf<<<gFeatT, blkT, 0, stream>>>(xbf, xT, NN, DD);

  // --- embed block 1 (aggregation shared with assign block 1) ---
  k_sage_agg<<<gStrip, blk256, 0, stream>>>(adjT, xT, dinv, aggE1);
  k_sage_lin<<<gStrip, blk256, 0, stream>>>(aggE1, xbf, Wb[0], be1, Wb[1], h1bf, nullptr);

  // --- embed block 2 ---
  k_trans_bf<<<gFeatT, blkT, 0, stream>>>(h1bf, h1T, NN, DD);
  k_sage_agg<<<gStrip, blk256, 0, stream>>>(adjT, h1T, dinv, aggT);
  k_sage_lin<<<gStrip, blk256, 0, stream>>>(aggT, h1bf, Wb[2], be2, Wb[3], hembbf, nullptr);
  k_trans_bf<<<gFeatT, blkT, 0, stream>>>(hembbf, hembT, NN, DD);

  // --- assign block 1 (reuses aggE1) + block 2 ---
  k_sage_lin<<<gStrip, blk256, 0, stream>>>(aggE1, xbf, Wb[4], ba1, Wb[5], a1bf, nullptr);
  k_trans_bf<<<gFeatT, blkT, 0, stream>>>(a1bf, a1T, NN, DD);
  k_sage_agg<<<gStrip, blk256, 0, stream>>>(adjT, a1T, dinv, aggT);
  k_sage_lin<<<gStrip, blk256, 0, stream>>>(aggT, a1bf, Wb[6], ba2, Wb[7], a2bf, a2f);

  // --- softmax + entropy, S^T ---
  k_softmax<<<BB * NN / 8, blk256, 0, stream>>>(a2f, Sbf, accum + 1);
  k_trans_bf<<<gFeatT, blkT, 0, stream>>>(Sbf, ST, NN, DD);

  // --- pooling ---
  k_gemm128<<<BB, blk256, 0, stream>>>(ST, hembT, out);                 // h_pooled
  k_pool_p<<<dim3(16, BB), blk256, 0, stream>>>(ST, adjT, Pbf);         // S^T adj
  k_gemm128<<<BB, blk256, 0, stream>>>(Pbf, ST, out + BB * DD * DD);    // adj_pooled

  // --- losses ---
  k_link<<<dim3(16, 16, BB), blk256, 0, stream>>>(Sbf, adj, accum);
  k_final<<<1, 1, 0, stream>>>(accum, out);
}
